// SelfInteractor_71519795413650
// MI455X (gfx1250) — matrix operand order repaired and checked by reading, not compiled, not run on hardware
//
#include <hip/hip_runtime.h>
#include <hip/hip_bf16.h>

// ---------------------------------------------------------------------------
// SelfInteractor for MI455X (gfx1250, wave32, WMMA + TDM).
//   N=1024 tokens, D=128.  ~0.8 GFLOP, ~10MB working set -> L2 resident.
//   q/k: v_wmma_f32_16x16x4_f32;  att/gi: v_wmma_f32_16x16x32_bf16;
//   att B/A tiles staged via tensor_load_to_lds (TDM) with bank-conflict-free
//   padded LDS layout; tanh/sigmoid via hardware v_exp_f32; GRU scan keeps
//   w_hh register-resident (192KB in VGPRs across 12 waves).
// ---------------------------------------------------------------------------

#define NTOK 1024
#define DDIM 128
#define TD3  384          // 3*D
#define NEG_INF -1e30f
#define LOG2E 1.4426950408889634f

typedef __attribute__((ext_vector_type(2)))  float        v2f;
typedef __attribute__((ext_vector_type(8)))  float        v8f;
typedef __attribute__((ext_vector_type(16))) __bf16       v16bf;
typedef __attribute__((ext_vector_type(8)))  unsigned int v8u;
typedef __attribute__((ext_vector_type(4)))  unsigned int v4ui;
typedef __attribute__((ext_vector_type(8)))  int          v8i;
typedef __attribute__((ext_vector_type(4)))  int          v4i;

__device__ __forceinline__ unsigned short f2bf(float f) {
  unsigned int b = __builtin_bit_cast(unsigned int, f);
  return (unsigned short)((b + 0x7FFFu + ((b >> 16) & 1u)) >> 16);
}
__device__ __forceinline__ float fast_tanh(float x) {
  x = fminf(10.f, fmaxf(-10.f, x));
  float e = __builtin_amdgcn_exp2f(x * (2.f * LOG2E));   // exp(2x)
  return (e - 1.f) * __builtin_amdgcn_rcpf(e + 1.f);
}
__device__ __forceinline__ float fast_sigmoid(float x) {
  float e = __builtin_amdgcn_exp2f(-x * LOG2E);
  return __builtin_amdgcn_rcpf(1.f + e);
}

// ---- TDM: 2-D bf16 tile load, global -> LDS (D# per ISA cdna5 §8) ---------
// Pads 1 dword after every 16 dwords (one 32-elem row) -> LDS row stride of
// 17 dwords, coprime with 64 banks => conflict-free fragment reads.
// Toolchain here is the 6-arg builtin: (u32x4, i32x8, i32x4, i32x4, i32x8, cpol).
__device__ __forceinline__ void tdm_load_2d_bf16(
    unsigned lds_off, const void* gaddr,
    unsigned tile_d0, unsigned tile_d1,
    unsigned tens_d0, unsigned tens_d1, unsigned stride_elems) {
  unsigned long long ga = (unsigned long long)(size_t)gaddr;
  v4ui g0;
  g0[0] = 1u;                                        // count=1, user mode
  g0[1] = lds_off;                                   // lds_addr
  g0[2] = (unsigned)(ga & 0xFFFFFFFFu);              // global_addr[31:0]
  g0[3] = (unsigned)((ga >> 32) & 0x01FFFFFFu)       // global_addr[56:32]
          | (2u << 30);                              // type = 2 (image)
  v8i g1;
  g1[0] = (int)((1u << 16)                           // data_size = 2 bytes
              | (1u << 20)                           // pad_enable
              | (3u << 22));                         // pad_interval = 16 dwords
                                                     // pad_amount = 0 => 1 dword
  g1[1] = (int)(tens_d0 << 16);                      // tensor_dim0 low16
  g1[2] = (int)((tens_d0 >> 16) | (tens_d1 << 16));  // dim0 hi | dim1 lo
  g1[3] = (int)((tens_d1 >> 16) | (tile_d0 << 16));  // dim1 hi | tile_dim0
  g1[4] = (int)(tile_d1 & 0xFFFFu);                  // tile_dim1 (tile_dim2=0)
  g1[5] = (int)stride_elems;                         // tensor_dim0_stride lo32
  g1[6] = 0;                                         // stride hi | dim1_stride lo
  g1[7] = 0;
  v4i z4 = {0, 0, 0, 0};
  v8i z8 = {0, 0, 0, 0, 0, 0, 0, 0};
  __builtin_amdgcn_tensor_load_to_lds(g0, g1, z4, z4, z8, 0);
}

// --------------------------- bf16 prep --------------------------------------
__global__ void prep_kernel(const float* __restrict__ X,
                            const float* __restrict__ w_ih,
                            unsigned short* __restrict__ Xb,
                            unsigned short* __restrict__ XT,
                            unsigned short* __restrict__ w_ihb) {
  int i = blockIdx.x * 256 + threadIdx.x;           // 512*256 = N*D threads
  if (i < NTOK * DDIM) {
    unsigned short h = f2bf(X[i]);
    Xb[i] = h;
    int r = i / DDIM, c = i % DDIM;
    XT[c * NTOK + r] = h;
  }
  if (i < TD3 * 2 * DDIM) w_ihb[i] = f2bf(w_ih[i]);
}

// --------------------------- q,k = X @ W^T  (fp32 WMMA) ----------------------
// grid (64,8,2), block 32 (one wave). 32 x v_wmma_f32_16x16x4_f32 per wave.
__global__ void qk_kernel(const float* __restrict__ X,
                          const float* __restrict__ Wq,
                          const float* __restrict__ Wk,
                          float* __restrict__ q, float* __restrict__ k) {
  const int lane = threadIdx.x;
  const int m  = lane & 15;
  const int hi = lane >> 4;
  const float* W  = blockIdx.z ? Wk : Wq;
  float*      out = blockIdx.z ? k  : q;
  const int arow = blockIdx.x * 16 + m;
  const int ncol = blockIdx.y * 16 + m;
  v8f c = {};
  for (int kt = 0; kt < DDIM; kt += 4) {
    int k0 = kt + hi * 2;
    v2f a = *(const v2f*)(X + arow * DDIM + k0);
    v2f b = *(const v2f*)(W + ncol * DDIM + k0);
    c = __builtin_amdgcn_wmma_f32_16x16x4_f32(false, a, false, b,
                                              (short)0, c, false, false);
  }
#pragma unroll
  for (int r = 0; r < 8; ++r) {
    int orow = blockIdx.x * 16 + r + hi * 8;
    out[orow * DDIM + m + blockIdx.y * 16] = c[r];
  }
}

// --------------------------- scores[i,j] = v . tanh(q_i + k_j) ---------------
__global__ void scores_kernel(const float* __restrict__ q,
                              const float* __restrict__ k,
                              const float* __restrict__ v,
                              float* __restrict__ S) {
  __shared__ float qs[DDIM], vs[DDIM];
  const int i = blockIdx.x;
  for (int d = threadIdx.x; d < DDIM; d += 256) { qs[d] = q[i * DDIM + d]; vs[d] = v[d]; }
  __syncthreads();
  for (int j = threadIdx.x; j < NTOK; j += 256) {
    float s = NEG_INF;
    if (j >= i) {
      const float* kr = k + j * DDIM;
      __builtin_prefetch(kr + 256 * DDIM, 0, 1);       // global_prefetch next tile
      s = 0.f;
#pragma unroll 4
      for (int d = 0; d < DDIM; ++d) s += vs[d] * fast_tanh(qs[d] + kr[d]);
    }
    S[i * NTOK + j] = s;
  }
}

// --------------------------- row softmax -> bf16 weights ---------------------
__global__ void softmax_kernel(const float* __restrict__ S,
                               unsigned short* __restrict__ Wb) {
  __shared__ float red[256];
  const int i = blockIdx.x, tid = threadIdx.x;
  float m = NEG_INF;
  float sv[4];
#pragma unroll
  for (int t = 0; t < 4; ++t) { sv[t] = S[i * NTOK + tid + t * 256]; m = fmaxf(m, sv[t]); }
  red[tid] = m; __syncthreads();
  for (int s = 128; s > 0; s >>= 1) { if (tid < s) red[tid] = fmaxf(red[tid], red[tid + s]); __syncthreads(); }
  const float mx = red[0]; __syncthreads();
  float e[4], sum = 0.f;
#pragma unroll
  for (int t = 0; t < 4; ++t) { e[t] = __builtin_amdgcn_exp2f((sv[t] - mx) * LOG2E); sum += e[t]; }
  red[tid] = sum; __syncthreads();
  for (int s = 128; s > 0; s >>= 1) { if (tid < s) red[tid] += red[tid + s]; __syncthreads(); }
  const float rs = __builtin_amdgcn_rcpf(red[0]);
#pragma unroll
  for (int t = 0; t < 4; ++t) Wb[i * NTOK + tid + t * 256] = f2bf(e[t] * rs);
}

// --------------------------- att = W @ X  (TDM -> LDS -> bf16 WMMA) ----------
// grid 64 blocks (one 16-row M tile each) x 256 threads (8 waves, one 16-col
// N tile per wave). Per K-step: wave 0 issues two tensor_load_to_lds DMAs
// (A: 16x32 of Wb, B: 128x32 of XT), waits TENSORcnt, barrier, then all waves
// build fragments from padded LDS (17-dword row stride, conflict-free).
__global__ void __launch_bounds__(256)
att_kernel(const unsigned short* __restrict__ Wb,
           const unsigned short* __restrict__ XT,
           unsigned short* __restrict__ attb) {
  __shared__ __align__(16) unsigned int ldsA[16 * 17];    // 16 rows (M) x 32 bf16 + pad
  __shared__ __align__(16) unsigned int ldsB[128 * 17];   // 128 cols (N) x 32 bf16 + pad
  const int lane = threadIdx.x & 31;
  const int wv   = threadIdx.x >> 5;                      // N-tile index 0..7
  const int m = lane & 15, hi = lane >> 4;
  const unsigned ldsA_off = (unsigned)(size_t)(void*)ldsA;
  const unsigned ldsB_off = (unsigned)(size_t)(void*)ldsB;
  v8f c = {};
  for (int kt = 0; kt < NTOK; kt += 32) {
    __syncthreads();                                      // prior reads done
    if (threadIdx.x < 32) {                               // wave 0 drives the TDM
      // A tile: rows [blockIdx.x*16, +16), k [kt, kt+32) of Wb (1024x1024)
      tdm_load_2d_bf16(ldsA_off, Wb + (size_t)blockIdx.x * 16 * NTOK + kt,
                       /*tile*/ 32, 16, /*tensor*/ NTOK, NTOK, NTOK);
      // B tile: cols [0,128), k [kt, kt+32) of XT (128 x 1024, k contiguous)
      tdm_load_2d_bf16(ldsB_off, XT + kt,
                       /*tile*/ 32, 128, /*tensor*/ NTOK, DDIM, NTOK);
      __builtin_amdgcn_s_wait_tensorcnt(0);
    }
    __syncthreads();                                      // LDS visible to all
    v8u a, b;
#pragma unroll
    for (int vv = 0; vv < 8; ++vv) {
      // A bf16 16x32 fragment: elem k = (vv<4?0:16)+hi*8+2*(vv&3) -> dword k/2
      a[vv] = ldsA[m * 17 + ((vv < 4) ? 0 : 8) + hi * 4 + (vv & 3)];
      // B bf16 32x16 fragment: elem k = (vv<4?0:16)+hi*8+2*(vv&3) -> dword k/2
      b[vv] = ldsB[(wv * 16 + m) * 17 + ((vv < 4) ? 0 : 8) + hi * 4 + (vv & 3)];
    }
    c = __builtin_amdgcn_wmma_f32_16x16x32_bf16(false, __builtin_bit_cast(v16bf, a),
                                                false, __builtin_bit_cast(v16bf, b),
                                                (short)0, c, false, false);
  }
#pragma unroll
  for (int r = 0; r < 8; ++r) {
    int orow = blockIdx.x * 16 + r + hi * 8;
    attb[orow * DDIM + wv * 16 + m] = f2bf(c[r]);
  }
}

// --------------------------- gi = [X,att] @ w_ih^T + b_ih  (bf16 WMMA) -------
__global__ void gi_kernel(const unsigned short* __restrict__ Xb,
                          const unsigned short* __restrict__ attb,
                          const unsigned short* __restrict__ w_ihb,
                          const float* __restrict__ b_ih,
                          float* __restrict__ gi) {
  const int lane = threadIdx.x;
  const int m = lane & 15, hi = lane >> 4;
  const int arow = blockIdx.x * 16 + m;
  const int ncol = blockIdx.y * 16 + m;
  v8f c = {};
  for (int kt = 0; kt < 2 * DDIM; kt += 32) {
    const unsigned short* Abase =
        (kt < DDIM) ? (Xb + arow * DDIM + kt) : (attb + arow * DDIM + (kt - DDIM));
    v8u a, b;
#pragma unroll
    for (int vv = 0; vv < 8; ++vv) {
      int ka = ((vv < 4) ? 0 : 16) + hi * 8 + 2 * (vv & 3);
      a[vv] = *(const unsigned int*)(Abase + ka);
      int kb = ((vv < 4) ? 0 : 16) + hi * 8 + 2 * (vv & 3);  // B[k][n] = w_ih[n][k], same k order as A
      b[vv] = *(const unsigned int*)(w_ihb + ncol * 2 * DDIM + kt + kb);
    }
    c = __builtin_amdgcn_wmma_f32_16x16x32_bf16(false, __builtin_bit_cast(v16bf, a),
                                                false, __builtin_bit_cast(v16bf, b),
                                                (short)0, c, false, false);
  }
#pragma unroll
  for (int r = 0; r < 8; ++r) {
    int orow = blockIdx.x * 16 + r + hi * 8;
    int ocol = blockIdx.y * 16 + m;
    gi[orow * TD3 + ocol] = c[r] + b_ih[ocol];
  }
}

// --------------------------- sequential GRU scan -----------------------------
__global__ void __launch_bounds__(TD3)
gru_kernel(const float* __restrict__ gi,
           const float* __restrict__ w_hh,
           const float* __restrict__ b_hh,
           float* __restrict__ out) {
  __shared__ float h[DDIM];
  __shared__ float g[TD3];
  const int idx = threadIdx.x;                 // 0..383
  float w[DDIM];
#pragma unroll
  for (int kk = 0; kk < DDIM; ++kk) w[kk] = w_hh[idx * DDIM + kk];
  const float bias = b_hh[idx];
  if (idx < DDIM) h[idx] = 0.f;
  __syncthreads();
  for (int t = 0; t < NTOK; ++t) {
    float acc = bias;
#pragma unroll
    for (int kk = 0; kk < DDIM; ++kk) acc += w[kk] * h[kk];   // LDS broadcast reads
    g[idx] = acc;                                             // gh[idx]
    __syncthreads();                                          // g ready, h reads done
    if (idx < DDIM) {
      const float* git = gi + t * TD3;
      float r   = fast_sigmoid(git[idx]            + g[idx]);
      float z   = fast_sigmoid(git[DDIM + idx]     + g[DDIM + idx]);
      float nst = fast_tanh   (git[2 * DDIM + idx] + r * g[2 * DDIM + idx]);
      float hn  = (1.f - z) * nst + z * h[idx];
      h[idx] = hn;
      out[t * DDIM + idx] = hn;
    }
    __syncthreads();                                          // h updated
  }
}

// ---------------------------------------------------------------------------
extern "C" void kernel_launch(void* const* d_in, const int* in_sizes, int n_in,
                              void* d_out, int out_size, void* d_ws, size_t ws_size,
                              hipStream_t stream) {
  const float* X    = (const float*)d_in[0];
  const float* Wq   = (const float*)d_in[1];
  const float* Wk   = (const float*)d_in[2];
  const float* v    = (const float*)d_in[3];
  const float* w_ih = (const float*)d_in[4];
  const float* w_hh = (const float*)d_in[5];
  const float* b_ih = (const float*)d_in[6];
  const float* b_hh = (const float*)d_in[7];
  float* out = (float*)d_out;

  // workspace carve-up (bytes)
  char* ws = (char*)d_ws;
  float*          q     = (float*)(ws + 0);                  // 512 KB
  float*          k     = (float*)(ws + (512u << 10));       // 512 KB
  float*          S     = (float*)(ws + (1024u << 10));      // 4 MB
  unsigned short* Wb    = (unsigned short*)(ws + (5120u << 10));  // 2 MB
  unsigned short* Xb    = (unsigned short*)(ws + (7168u << 10));  // 256 KB
  unsigned short* XT    = (unsigned short*)(ws + (7424u << 10));  // 256 KB
  unsigned short* attb  = (unsigned short*)(ws + (7680u << 10));  // 256 KB
  unsigned short* w_ihb = (unsigned short*)(ws + (7936u << 10));  // 192 KB
  float*          gi    = (float*)(ws + (8128u << 10));      // 1.5 MB -> ~9.6MB total

  prep_kernel   <<<dim3((NTOK * DDIM) / 256), 256, 0, stream>>>(X, w_ih, Xb, XT, w_ihb);
  qk_kernel     <<<dim3(NTOK / 16, DDIM / 16, 2), 32, 0, stream>>>(X, Wq, Wk, q, k);
  scores_kernel <<<dim3(NTOK), 256, 0, stream>>>(q, k, v, S);
  softmax_kernel<<<dim3(NTOK), 256, 0, stream>>>(S, Wb);
  att_kernel    <<<dim3(NTOK / 16), 256, 0, stream>>>(Wb, XT, attb);
  gi_kernel     <<<dim3(NTOK / 16, TD3 / 16), 32, 0, stream>>>(Xb, attb, w_ihb, b_ih, gi);
  gru_kernel    <<<dim3(1), TD3, 0, stream>>>(gi, w_hh, b_hh, out);
}
